// Mamba2StepCore_73572789781062
// MI455X (gfx1250) — matrix operand order repaired and checked by reading, not compiled, not run on hardware
//
#include <hip/hip_runtime.h>
#include <hip/hip_bf16.h>
#include <math.h>

// ---------------- problem constants ----------------
#define B_SZ      512
#define INPUT_DIM 1536
#define MAMBA_DIM 1024
#define OUT_DIM   1024
#define D_STATE   128
#define D_CONV    4
#define HEADDIM   64
#define D_INNER   2048
#define NHEADS    32
#define CONV_DIM  2304          // D_INNER + 2*D_STATE
#define D_IN_PROJ 4384          // 2*D_INNER + 2*D_STATE + NHEADS
#define EPS       1e-5f

// ---------------- vector / WMMA types ----------------
typedef float        f4    __attribute__((ext_vector_type(4)));
typedef unsigned int u4    __attribute__((ext_vector_type(4)));
typedef unsigned int u2    __attribute__((ext_vector_type(2)));
typedef __bf16       v16bf __attribute__((ext_vector_type(16)));
typedef float        v8f   __attribute__((ext_vector_type(8)));

union Frag { u4 q[2]; v16bf v; };

__device__ __forceinline__ float silu_f(float x) { return x / (1.f + __expf(-x)); }

// =====================================================================
// bf16 WMMA GEMM:  C[M,N] = A[M,K] @ W[K,N] (+bias), fp32 in/out,
// fp32->bf16 conversion during LDS staging, fp32 accumulation.
// Block tile 128x128, BK=32, 256 threads (8 waves, 4Mx2N),
// wave tile 32x64 -> 8 WMMAs per K-step. Double-buffered LDS,
// software-pipelined global fetch, one barrier per K-step.
// =====================================================================
#define BM  128
#define BN  128
#define BK  32
#define LDA 40   // padded LDS row stride (bf16 elems) -> 80B, 16B aligned
#define LDB 40

// fetch next K-tile from global into registers
__device__ __forceinline__ void stage_fetch(const float* __restrict__ A,
                                            const float* __restrict__ W,
                                            int N, int K, int m0, int n0, int k0,
                                            int tid, f4 ra[4], float rb[16])
{
    #pragma unroll
    for (int i = 0; i < 4; ++i) {               // A: 128x32 fp32, 4 x b128 / thread
        int idx = tid + i * 256;
        int r   = idx >> 3;
        int c4  = (idx & 7) * 4;
        ra[i] = *(const f4*)(A + (size_t)(m0 + r) * K + k0 + c4);
    }
    // B: 32x128 fp32; thread owns column n = tid&127, k-range (tid>>7)*16..+16.
    // Lanes have consecutive n -> coalesced 128B per k row.
    int n     = tid & 127;
    int khalf = (tid >> 7) * 16;
    int gn    = n0 + n;
    bool ok   = gn < N;
    #pragma unroll
    for (int g = 0; g < 4; ++g) {
        int k = khalf + g * 4;
        #pragma unroll
        for (int j = 0; j < 4; ++j)
            rb[g * 4 + j] = ok ? W[(size_t)(k0 + k + j) * N + gn] : 0.f;
    }
}

// convert + store staged registers into an LDS buffer
__device__ __forceinline__ void stage_commit(__bf16* __restrict__ sA,
                                             __bf16* __restrict__ sB,
                                             int tid, const f4 ra[4], const float rb[16])
{
    #pragma unroll
    for (int i = 0; i < 4; ++i) {
        int idx = tid + i * 256;
        int r   = idx >> 3;
        int c4  = (idx & 7) * 4;
        __bf16* d = &sA[r * LDA + c4];
        d[0] = (__bf16)ra[i].x; d[1] = (__bf16)ra[i].y;
        d[2] = (__bf16)ra[i].z; d[3] = (__bf16)ra[i].w;
    }
    int n     = tid & 127;
    int khalf = (tid >> 7) * 16;
    #pragma unroll
    for (int g = 0; g < 4; ++g) {               // 4 x ds_store_b64, k-contiguous
        __attribute__((aligned(8))) __bf16 t[4];
        t[0] = (__bf16)rb[g * 4 + 0]; t[1] = (__bf16)rb[g * 4 + 1];
        t[2] = (__bf16)rb[g * 4 + 2]; t[3] = (__bf16)rb[g * 4 + 3];
        *(u2*)&sB[n * LDB + khalf + g * 4] = *(const u2*)t;
    }
}

__global__ __launch_bounds__(256)
void gemm_bf16_wmma(const float* __restrict__ A, const float* __restrict__ W,
                    float* __restrict__ C, const float* __restrict__ bias,
                    int M, int N, int K)
{
    __shared__ __bf16 sA[2][BM * LDA];     // [row m][k]
    __shared__ __bf16 sB[2][BN * LDB];     // transposed: [col n][k]

    const int tid  = threadIdx.x;
    const int lane = tid & 31;
    const int wv   = tid >> 5;             // 0..7
    const int waveM = (wv & 3) * 32;       // 0,32,64,96
    const int waveN = (wv >> 2) * 64;      // 0,64
    const int m0 = blockIdx.y * BM;
    const int n0 = blockIdx.x * BN;

    v8f acc[2][4];
    #pragma unroll
    for (int i = 0; i < 2; ++i)
        #pragma unroll
        for (int j = 0; j < 4; ++j)
            #pragma unroll
            for (int e = 0; e < 8; ++e) acc[i][j][e] = 0.f;

    f4    ra[4];
    float rb[16];

    // prologue: stage K-tile 0
    stage_fetch(A, W, N, K, m0, n0, 0, tid, ra, rb);
    stage_commit(sA[0], sB[0], tid, ra, rb);
    __syncthreads();

    const int nk = K / BK;
    const int mrow = lane & 15;
    const int aoff = (lane < 16) ? 0 : 16;    // A chunk1 byte offset
    const int boff = (lane < 16) ? 0 : 16;    // B chunk1 byte offset

    for (int kb = 0; kb < nk; ++kb) {
        const int cur = kb & 1;
        if (kb + 1 < nk)   // prefetch next tile while matrix pipe works
            stage_fetch(A, W, N, K, m0, n0, (kb + 1) * BK, tid, ra, rb);

        // ---- fragments per ISA VGPR layouts ----
        // A 16x32 bf16: lanes 0-15 -> K {0..7,16..23}; lanes 16-31 -> K {8..15,24..31}
        Frag fa[2], fb[4];
        #pragma unroll
        for (int mi = 0; mi < 2; ++mi) {
            const char* p = (const char*)&sA[cur][(waveM + mi * 16 + mrow) * LDA] + aoff;
            fa[mi].q[0] = *(const u4*)(p);
            fa[mi].q[1] = *(const u4*)(p + 32);
        }
        // B 32x16 bf16: lane n holds column n; K {0..7,16..23} / {8..15,24..31} by half-wave
        #pragma unroll
        for (int ni = 0; ni < 4; ++ni) {
            const char* p = (const char*)&sB[cur][(waveN + ni * 16 + mrow) * LDB] + boff;
            fb[ni].q[0] = *(const u4*)(p);
            fb[ni].q[1] = *(const u4*)(p + 32);
        }

        #pragma unroll
        for (int mi = 0; mi < 2; ++mi)
            #pragma unroll
            for (int ni = 0; ni < 4; ++ni)
                acc[mi][ni] = __builtin_amdgcn_wmma_f32_16x16x32_bf16(
                    false, fa[mi].v, false, fb[ni].v,
                    (short)0, acc[mi][ni], false, false);

        if (kb + 1 < nk)
            stage_commit(sA[cur ^ 1], sB[cur ^ 1], tid, ra, rb);
        __syncthreads();
    }

    // ---- epilogue: C/D layout lane n = lane%16, M = vgpr + 8*(lane>=16) ----
    const int ncol = lane & 15;
    const int mo   = (lane >= 16) ? 8 : 0;
    #pragma unroll
    for (int mi = 0; mi < 2; ++mi)
        #pragma unroll
        for (int ni = 0; ni < 4; ++ni) {
            int gn = n0 + waveN + ni * 16 + ncol;
            if (gn < N) {
                float bv = bias ? bias[gn] : 0.f;
                #pragma unroll
                for (int v = 0; v < 8; ++v) {
                    int gm = m0 + waveM + mi * 16 + mo + v;
                    C[(size_t)gm * N + gn] = acc[mi][ni][v] + bv;
                }
            }
        }
}

// =====================================================================
// Row RMS-norm (+optional SiLU): one block per row of 1024, 256 thr.
// =====================================================================
__global__ __launch_bounds__(256)
void rmsnorm_kernel(const float* __restrict__ in, const float* __restrict__ g,
                    float* __restrict__ out, int ncols, int do_silu)
{
    const int row = blockIdx.x;
    const int tid = threadIdx.x;
    const float* r = in + (size_t)row * ncols;
    float v[4];
    float ss = 0.f;
    #pragma unroll
    for (int i = 0; i < 4; ++i) {
        v[i] = r[tid + i * 256];
        ss += v[i] * v[i];
    }
    #pragma unroll
    for (int off = 16; off > 0; off >>= 1) ss += __shfl_down(ss, off, 32);
    __shared__ float red[8];
    if ((tid & 31) == 0) red[tid >> 5] = ss;
    __syncthreads();
    if (tid == 0) {
        float t = 0.f;
        #pragma unroll
        for (int i = 0; i < 8; ++i) t += red[i];
        red[0] = t;
    }
    __syncthreads();
    const float rs = rsqrtf(red[0] / (float)ncols + EPS);
    float* o = out + (size_t)row * ncols;
    #pragma unroll
    for (int i = 0; i < 4; ++i) {
        int c = tid + i * 256;
        float y = v[i] * rs * g[c];
        o[c] = do_silu ? silu_f(y) : y;
    }
}

// =====================================================================
// Conv-state shift + depthwise conv + SiLU.
// Writes conv_state_new into output slab; xbc_act to workspace.
// =====================================================================
__global__ __launch_bounds__(256)
void conv_kernel(const float* __restrict__ zxbcdt, const float* __restrict__ conv_state,
                 const float* __restrict__ conv_w, const float* __restrict__ conv_b,
                 float* __restrict__ conv_out, float* __restrict__ xbc_act)
{
    const int ch = blockIdx.x * 256 + threadIdx.x;      // 0..CONV_DIM-1
    const int b  = blockIdx.y;
    if (ch >= CONV_DIM) return;
    const size_t idx = (size_t)b * CONV_DIM + ch;

    f4 cs = *(const f4*)(conv_state + idx * D_CONV);
    float xin = zxbcdt[(size_t)b * D_IN_PROJ + D_INNER + ch];
    f4 ns; ns.x = cs.y; ns.y = cs.z; ns.z = cs.w; ns.w = xin;
    *(f4*)(conv_out + idx * D_CONV) = ns;

    f4 w = *(const f4*)(conv_w + (size_t)ch * D_CONV);
    float val = ns.x * w.x + ns.y * w.y + ns.z * w.z + ns.w * w.w + conv_b[ch];
    xbc_act[idx] = silu_f(val);
}

// =====================================================================
// dt = softplus(dt_raw + dt_bias); da = exp(dt * -exp(A_log))
// =====================================================================
__global__ __launch_bounds__(256)
void dtda_kernel(const float* __restrict__ zxbcdt, const float* __restrict__ dt_bias,
                 const float* __restrict__ A_log,
                 float* __restrict__ dt_out, float* __restrict__ da_out)
{
    const int idx = blockIdx.x * 256 + threadIdx.x;     // b*NHEADS + h
    if (idx >= B_SZ * NHEADS) return;
    const int b = idx >> 5;
    const int h = idx & 31;
    float d = zxbcdt[(size_t)b * D_IN_PROJ + (D_INNER + CONV_DIM) + h] + dt_bias[h];
    float sp = (d > 20.f) ? d : log1pf(__expf(d));
    dt_out[idx] = sp;
    da_out[idx] = __expf(sp * (-__expf(A_log[h])));
}

// =====================================================================
// SSM state update (memory-bound, NT streaming) + y epilogue.
// One block per (head, batch): 256 thr, thread = (p = tid>>2, quad of 32 n's).
// 512MB in + 512MB out of fp32 state -> the roofline term (~46us @23.3TB/s).
// =====================================================================
__global__ __launch_bounds__(256)
void ssm_kernel(const float* __restrict__ ssm_state, const float* __restrict__ zxbcdt,
                const float* __restrict__ xbc_act, const float* __restrict__ dt_arr,
                const float* __restrict__ da_arr, const float* __restrict__ Dvec,
                float* __restrict__ ssm_out, float* __restrict__ ym)
{
    const int h = blockIdx.x;
    const int b = blockIdx.y;
    const int tid = threadIdx.x;

    __shared__ float sh_xh[HEADDIM];
    __shared__ float sh_b[D_STATE];
    __shared__ float sh_c[D_STATE];
    const float* xrow = xbc_act + (size_t)b * CONV_DIM;
    if (tid < HEADDIM)  sh_xh[tid] = xrow[h * HEADDIM + tid];
    if (tid < D_STATE)  sh_b[tid]  = xrow[D_INNER + tid];
    else if (tid < 2 * D_STATE) sh_c[tid - D_STATE] = xrow[D_INNER + D_STATE + (tid - D_STATE)];
    __syncthreads();

    const float dt = dt_arr[b * NHEADS + h];
    const float da = da_arr[b * NHEADS + h];

    const int p  = tid >> 2;
    const int q  = tid & 3;
    const int n0 = q * 32;
    const float xp   = sh_xh[p];
    const float coef = dt * xp;

    const size_t off = (((size_t)b * NHEADS + h) * HEADDIM + p) * D_STATE + n0;
    const f4* sp = (const f4*)(ssm_state + off);
    f4*       dp = (f4*)(ssm_out + off);

    float acc = 0.f;
    #pragma unroll
    for (int i = 0; i < 8; ++i) {
        f4 s  = __builtin_nontemporal_load(sp + i);       // stream in
        f4 bb = *(const f4*)&sh_b[n0 + i * 4];
        f4 cc = *(const f4*)&sh_c[n0 + i * 4];
        f4 sn = s * da + coef * bb;
        __builtin_nontemporal_store(sn, dp + i);          // stream out
        f4 t = sn * cc;
        acc += t.x + t.y + t.z + t.w;
    }
    acc += __shfl_xor(acc, 1, 32);
    acc += __shfl_xor(acc, 2, 32);

    if (q == 0) {
        float y = acc + Dvec[h] * xp;
        float z = zxbcdt[(size_t)b * D_IN_PROJ + h * HEADDIM + p];
        ym[(size_t)b * D_INNER + h * HEADDIM + p] = y * silu_f(z);
    }
}

// =====================================================================
// Host-side orchestration
// =====================================================================
extern "C" void kernel_launch(void* const* d_in, const int* in_sizes, int n_in,
                              void* d_out, int out_size, void* d_ws, size_t ws_size,
                              hipStream_t stream)
{
    const float* x          = (const float*)d_in[0];
    const float* conv_state = (const float*)d_in[1];
    const float* ssm_state  = (const float*)d_in[2];
    const float* w_in       = (const float*)d_in[3];
    const float* b_in       = (const float*)d_in[4];
    const float* g_in       = (const float*)d_in[5];
    const float* W_inproj   = (const float*)d_in[6];
    const float* conv_w     = (const float*)d_in[7];
    const float* conv_b     = (const float*)d_in[8];
    const float* A_log      = (const float*)d_in[9];
    const float* dt_bias    = (const float*)d_in[10];
    const float* Dvec       = (const float*)d_in[11];
    const float* W_mamba_out= (const float*)d_in[12];
    const float* w_out      = (const float*)d_in[13];
    const float* b_out      = (const float*)d_in[14];
    const float* g_outnorm  = (const float*)d_in[15];

    float* out = (float*)d_out;
    float* y_out    = out;                                   // [512,1024]
    float* conv_out = out + (size_t)B_SZ * OUT_DIM;          // [512,2304,4]
    float* ssm_out  = conv_out + (size_t)B_SZ * CONV_DIM * D_CONV; // [512,32,64,128]

    float* ws = (float*)d_ws;
    float* t1      = ws;                                      // [512,1024]
    float* h_act   = t1      + (size_t)B_SZ * MAMBA_DIM;      // [512,1024]
    float* zxbcdt  = h_act   + (size_t)B_SZ * MAMBA_DIM;      // [512,4384]
    float* xbc_act = zxbcdt  + (size_t)B_SZ * D_IN_PROJ;      // [512,2304]
    float* dt_arr  = xbc_act + (size_t)B_SZ * CONV_DIM;       // [512,32]
    float* da_arr  = dt_arr  + (size_t)B_SZ * NHEADS;         // [512,32]
    float* ym      = da_arr  + (size_t)B_SZ * NHEADS;         // [512,2048]
    float* t2      = ym      + (size_t)B_SZ * D_INNER;        // [512,1024]
    float* t3      = t2      + (size_t)B_SZ * MAMBA_DIM;      // [512,1024]

    dim3 blk(256);

    // 1) t1 = x @ w_in + b_in
    gemm_bf16_wmma<<<dim3(MAMBA_DIM / BN, B_SZ / BM), blk, 0, stream>>>(
        x, w_in, t1, b_in, B_SZ, MAMBA_DIM, INPUT_DIM);

    // 2) h = silu(rmsnorm(t1, g_in))
    rmsnorm_kernel<<<dim3(B_SZ), blk, 0, stream>>>(t1, g_in, h_act, MAMBA_DIM, 1);

    // 3) zxbcdt = h @ W_inproj   (N=4384, edge-guarded)
    gemm_bf16_wmma<<<dim3((D_IN_PROJ + BN - 1) / BN, B_SZ / BM), blk, 0, stream>>>(
        h_act, W_inproj, zxbcdt, nullptr, B_SZ, D_IN_PROJ, MAMBA_DIM);

    // 4) conv shift + depthwise conv + silu
    conv_kernel<<<dim3((CONV_DIM + 255) / 256, B_SZ), blk, 0, stream>>>(
        zxbcdt, conv_state, conv_w, conv_b, conv_out, xbc_act);

    // 5) dt/da
    dtda_kernel<<<dim3((B_SZ * NHEADS) / 256), blk, 0, stream>>>(
        zxbcdt, dt_bias, A_log, dt_arr, da_arr);

    // 6) SSM state stream + y epilogue (fuses +D*xh and *silu(z))
    ssm_kernel<<<dim3(NHEADS, B_SZ), blk, 0, stream>>>(
        ssm_state, zxbcdt, xbc_act, dt_arr, da_arr, Dvec, ssm_out, ym);

    // 7) t2 = ym @ W_mamba_out
    gemm_bf16_wmma<<<dim3(MAMBA_DIM / BN, B_SZ / BM), blk, 0, stream>>>(
        ym, W_mamba_out, t2, nullptr, B_SZ, MAMBA_DIM, D_INNER);

    // 8) t3 = t2 @ w_out + b_out
    gemm_bf16_wmma<<<dim3(OUT_DIM / BN, B_SZ / BM), blk, 0, stream>>>(
        t2, w_out, t3, b_out, B_SZ, OUT_DIM, MAMBA_DIM);

    // 9) y = rmsnorm(t3, g_outnorm)
    rmsnorm_kernel<<<dim3(B_SZ), blk, 0, stream>>>(t3, g_outnorm, y_out, OUT_DIM, 0);
}
